// LSTM_45810121179628
// MI455X (gfx1250) — matrix operand order repaired and checked by reading, not compiled, not run on hardware
//
#include <hip/hip_runtime.h>

// ---------------------------------------------------------------------------
// 8-layer LSTM (B=1024, T=64, IN=64, H=512) + FC(C=128) for MI455X (gfx1250).
// Matmuls via v_wmma_f32_16x16x32_bf16; tiles staged with CDNA5 async
// global->LDS copies (ASYNCcnt) and double-buffered LDS.
// ---------------------------------------------------------------------------

#define B_   1024
#define T_   64
#define IN_  64
#define H_   512
#define L_   8
#define C_   128
#define G4H  (4 * H_)

typedef __attribute__((ext_vector_type(16))) __bf16 v16bf;
typedef __attribute__((ext_vector_type(8)))  float  v8f;
typedef unsigned short u16;

static __device__ __forceinline__ u16 f2bf(float f) {
  union { float f; unsigned u; } v; v.f = f;
  unsigned r = v.u + 0x7FFFu + ((v.u >> 16) & 1u);   // round-to-nearest-even
  return (u16)(r >> 16);
}

union Frag { v16bf v; uint4 q[2]; };

// CDNA5 async copy: 16 bytes per lane, global -> LDS, tracked by ASYNCcnt.
// LDS operand = low 32 bits of the generic shared pointer (LDS aperture).
static __device__ __forceinline__ void async_copy16(const u16* g, u16* l) {
  unsigned lds = (unsigned)(unsigned long long)(const void*)l;
  asm volatile("global_load_async_to_lds_b128 %0, %1, off"
               :: "v"(lds), "v"(g) : "memory");
}

// ---------------------------------------------------------------------------
// WMMA GEMM:  C[M,N](f32) (+)= A[M,K](bf16,row-major,lda) @ B[N,K](bf16)^T
// Block tile 128x128, K-chunk 32, double-buffered LDS with async staging.
// 8 waves; each wave owns 64x32 (4x2 WMMA tiles). Shapes are exact multiples.
// ---------------------------------------------------------------------------
#define BM 128
#define BN 128
#define BK 32
#define LSTR (BK + 8)   // padded LDS row: 80B -> conflict-free b128 frag reads

__global__ __launch_bounds__(256)
void gemm_bf16_wmma(const u16* __restrict__ A, int lda,
                    const u16* __restrict__ Bm, int ldb,
                    float* __restrict__ C, int ldc,
                    int K, int accumulate)
{
  __shared__ __align__(16) u16 As[2][BM * LSTR];
  __shared__ __align__(16) u16 Bs[2][BN * LSTR];

  const int tid  = threadIdx.x;
  const int wave = tid >> 5;
  const int lane = tid & 31;
  const int hi   = lane >> 4;      // which 16-lane half (ISA frag layouts)
  const int ln   = lane & 15;

  const int bm = blockIdx.y * BM;
  const int bn = blockIdx.x * BN;
  const int wm = (wave >> 2) * 64; // 2 wave-rows in M
  const int wn = (wave & 3)  * 32; // 4 wave-cols in N

  // Per-thread staging coords: 4 async 16B chunks per stage (2 for A, 2 for B).
  const int rr = tid >> 2;         // 0..63
  const int kc = (tid & 3) * 8;    // 0,8,16,24

  v8f acc[4][2];
  if (accumulate) {
    #pragma unroll
    for (int mt = 0; mt < 4; ++mt)
      #pragma unroll
      for (int nt = 0; nt < 2; ++nt) {
        const int rbase = bm + wm + mt * 16 + hi * 8;  // C layout: VGPR r -> M=r(+8)
        const int col   = bn + wn + nt * 16 + ln;      // lanes 0-15 -> N
        #pragma unroll
        for (int r = 0; r < 8; ++r)
          acc[mt][nt][r] = C[(size_t)(rbase + r) * ldc + col];
      }
  } else {
    #pragma unroll
    for (int mt = 0; mt < 4; ++mt)
      #pragma unroll
      for (int nt = 0; nt < 2; ++nt)
        #pragma unroll
        for (int r = 0; r < 8; ++r)
          acc[mt][nt][r] = 0.0f;
  }

  // Issue one stage (4 async copies) for K-chunk k0 into LDS buffer `buf`.
  auto stage = [&](int buf, int k0) {
    async_copy16(A  + (size_t)(bm + rr)      * lda + k0 + kc, &As[buf][rr * LSTR + kc]);
    async_copy16(A  + (size_t)(bm + 64 + rr) * lda + k0 + kc, &As[buf][(64 + rr) * LSTR + kc]);
    async_copy16(Bm + (size_t)(bn + rr)      * ldb + k0 + kc, &Bs[buf][rr * LSTR + kc]);
    async_copy16(Bm + (size_t)(bn + 64 + rr) * ldb + k0 + kc, &Bs[buf][(64 + rr) * LSTR + kc]);
  };

  const int nk = K / BK;
  stage(0, 0);

  for (int i = 0; i < nk; ++i) {
    const int p = i & 1;
    if (i + 1 < nk) {
      stage(p ^ 1, (i + 1) * BK);              // overlap next-tile fetch
      asm volatile("s_wait_asynccnt 0x4" ::: "memory");  // 4 new pending -> stage i done
    } else {
      asm volatile("s_wait_asynccnt 0x0" ::: "memory");
    }
    __syncthreads();                           // all waves' buf-p data visible

    // A frag (16x32 bf16): lane = row; elems K = hi*8+0..7 then 16+hi*8+0..7.
    Frag af[4];
    #pragma unroll
    for (int mt = 0; mt < 4; ++mt) {
      const u16* src = As[p] + (wm + mt * 16 + ln) * LSTR;
      af[mt].q[0] = *(const uint4*)(src + hi * 8);
      af[mt].q[1] = *(const uint4*)(src + 16 + hi * 8);
    }
    // B frag (32x16 bf16): lane = col (row of B'); elems K = hi*8+0..7 then 16+hi*8+0..7.
    Frag bf[2];
    #pragma unroll
    for (int nt = 0; nt < 2; ++nt) {
      const u16* src = Bs[p] + (wn + nt * 16 + ln) * LSTR;
      bf[nt].q[0] = *(const uint4*)(src + hi * 8);
      bf[nt].q[1] = *(const uint4*)(src + 16 + hi * 8);
    }

    #pragma unroll
    for (int mt = 0; mt < 4; ++mt)
      #pragma unroll
      for (int nt = 0; nt < 2; ++nt)
        acc[mt][nt] = __builtin_amdgcn_wmma_f32_16x16x32_bf16(
            false, af[mt].v, false, bf[nt].v,
            (short)0, acc[mt][nt], false, false);

    __syncthreads();                           // all waves done reading buf p
  }

  #pragma unroll
  for (int mt = 0; mt < 4; ++mt)
    #pragma unroll
    for (int nt = 0; nt < 2; ++nt) {
      const int rbase = bm + wm + mt * 16 + hi * 8;
      const int col   = bn + wn + nt * 16 + ln;
      #pragma unroll
      for (int r = 0; r < 8; ++r)
        C[(size_t)(rbase + r) * ldc + col] = acc[mt][nt][r];
    }
}

// ---------------------------------------------------------------------------
// Pointwise LSTM cell: gates (f32) -> c,h. Writes bf16 h into the sequence
// buffer (consumed by next step / next layer) and optionally into the
// [B, T*H] FC operand buffer (last layer only).
// ---------------------------------------------------------------------------
__global__ __launch_bounds__(256)
void lstm_cell(const float* __restrict__ G,
               const float* __restrict__ bi, const float* __restrict__ bh,
               float* __restrict__ cst,
               u16* __restrict__ hout,
               u16* __restrict__ fcA, int t, int first)
{
  const int idx = blockIdx.x * blockDim.x + threadIdx.x;  // 0 .. B*H-1
  const int b = idx >> 9;
  const int h = idx & (H_ - 1);
  const float* g = G + (size_t)b * G4H;

  float ig = g[h]            + bi[h]            + bh[h];
  float fg = g[H_ + h]       + bi[H_ + h]       + bh[H_ + h];
  float gg = g[2 * H_ + h]   + bi[2 * H_ + h]   + bh[2 * H_ + h];
  float og = g[3 * H_ + h]   + bi[3 * H_ + h]   + bh[3 * H_ + h];

  float i = 1.0f / (1.0f + __expf(-ig));
  float f = 1.0f / (1.0f + __expf(-fg));
  float o = 1.0f / (1.0f + __expf(-og));
  float gt = tanhf(gg);

  float c = first ? 0.0f : cst[idx];
  c = f * c + i * gt;
  float hv = o * tanhf(c);

  cst[idx]  = c;
  hout[idx] = f2bf(hv);
  if (fcA) fcA[(size_t)b * (T_ * H_) + t * H_ + h] = f2bf(hv);
}

__global__ __launch_bounds__(256)
void cvt_f32_bf16(const float* __restrict__ s, u16* __restrict__ d, int n) {
  int i = blockIdx.x * blockDim.x + threadIdx.x;
  if (i < n) d[i] = f2bf(s[i]);
}

__global__ __launch_bounds__(256)
void add_bias(float* __restrict__ out, const float* __restrict__ b) {
  int i = blockIdx.x * blockDim.x + threadIdx.x;  // B_*C_ threads
  out[i] += b[i & (C_ - 1)];
}

// ---------------------------------------------------------------------------
extern "C" void kernel_launch(void* const* d_in, const int* in_sizes, int n_in,
                              void* d_out, int out_size, void* d_ws, size_t ws_size,
                              hipStream_t stream)
{
  const float* x     = (const float*)d_in[0];
  const float* w_ih0 = (const float*)d_in[1];
  const float* w_hh0 = (const float*)d_in[2];
  const float* b_ih0 = (const float*)d_in[3];
  const float* b_hh0 = (const float*)d_in[4];
  const float* w_ih  = (const float*)d_in[5];
  const float* w_hh  = (const float*)d_in[6];
  const float* b_ih  = (const float*)d_in[7];
  const float* b_hh  = (const float*)d_in[8];
  const float* fc_w  = (const float*)d_in[9];
  const float* fc_b  = (const float*)d_in[10];
  float* out = (float*)d_out;

  char* p = (char*)d_ws;
  auto carve = [&](size_t bytes) {
    char* r = p; p += (bytes + 255) & ~(size_t)255; return r;
  };
  u16*   xbf  = (u16*)  carve((size_t)B_ * T_ * IN_ * 2);        // x in bf16
  u16*   seq  = (u16*)  carve((size_t)T_ * B_ * H_ * 2);         // h sequence (in-place across layers)
  u16*   fcA  = (u16*)  carve((size_t)B_ * T_ * H_ * 2);         // [B, T*H] FC operand
  float* G    = (float*)carve((size_t)B_ * G4H * 4);             // gate pre-activations
  float* cs   = (float*)carve((size_t)B_ * H_ * 4);              // cell state
  u16*   wi0b = (u16*)  carve((size_t)G4H * IN_ * 2);
  u16*   wh0b = (u16*)  carve((size_t)G4H * H_ * 2);
  u16*   wib  = (u16*)  carve((size_t)(L_ - 1) * G4H * H_ * 2);
  u16*   whb  = (u16*)  carve((size_t)(L_ - 1) * G4H * H_ * 2);
  u16*   fwb  = (u16*)  carve((size_t)C_ * T_ * H_ * 2);

  const dim3 blk(256);
  auto CVT = [&](const float* s, u16* d, size_t n) {
    cvt_f32_bf16<<<dim3((unsigned)((n + 255) / 256)), blk, 0, stream>>>(s, d, (int)n);
  };
  CVT(x,     xbf,  (size_t)B_ * T_ * IN_);
  CVT(w_ih0, wi0b, (size_t)G4H * IN_);
  CVT(w_hh0, wh0b, (size_t)G4H * H_);
  CVT(w_ih,  wib,  (size_t)(L_ - 1) * G4H * H_);
  CVT(w_hh,  whb,  (size_t)(L_ - 1) * G4H * H_);
  CVT(fc_w,  fwb,  (size_t)C_ * T_ * H_);

  const dim3 gemm_grid(G4H / BN, B_ / BM);       // (16, 8)
  const dim3 cell_grid(B_ * H_ / 256);
  const size_t BH = (size_t)B_ * H_;

  // ---- Layer 0 (D_in = 64) ----
  for (int t = 0; t < T_; ++t) {
    gemm_bf16_wmma<<<gemm_grid, blk, 0, stream>>>(
        xbf + (size_t)t * IN_, T_ * IN_, wi0b, IN_, G, G4H, IN_, 0);
    if (t > 0)
      gemm_bf16_wmma<<<gemm_grid, blk, 0, stream>>>(
          seq + (size_t)(t - 1) * BH, H_, wh0b, H_, G, G4H, H_, 1);
    lstm_cell<<<cell_grid, blk, 0, stream>>>(
        G, b_ih0, b_hh0, cs, seq + (size_t)t * BH, nullptr, t, t == 0);
  }

  // ---- Layers 1..7 (in-place on seq; reads seq[t] before overwriting it) ----
  for (int l = 0; l < L_ - 1; ++l) {
    const u16*   wi_l = wib + (size_t)l * G4H * H_;
    const u16*   wh_l = whb + (size_t)l * G4H * H_;
    const float* bi_l = b_ih + (size_t)l * G4H;
    const float* bh_l = b_hh + (size_t)l * G4H;
    u16* fcw = (l == L_ - 2) ? fcA : nullptr;
    for (int t = 0; t < T_; ++t) {
      gemm_bf16_wmma<<<gemm_grid, blk, 0, stream>>>(
          seq + (size_t)t * BH, H_, wi_l, H_, G, G4H, H_, 0);
      if (t > 0)
        gemm_bf16_wmma<<<gemm_grid, blk, 0, stream>>>(
            seq + (size_t)(t - 1) * BH, H_, wh_l, H_, G, G4H, H_, 1);
      lstm_cell<<<cell_grid, blk, 0, stream>>>(
          G, bi_l, bh_l, cs, seq + (size_t)t * BH, fcw, t, t == 0);
    }
  }

  // ---- FC: out[B,128] = fcA[B, T*H] @ fc_w[128, T*H]^T + fc_b ----
  const dim3 fc_grid(C_ / BN, B_ / BM);          // (1, 8)
  gemm_bf16_wmma<<<fc_grid, blk, 0, stream>>>(
      fcA, T_ * H_, fwb, T_ * H_, out, C_, T_ * H_, 0);
  add_bias<<<dim3(B_ * C_ / 256), blk, 0, stream>>>(out, fc_b);
}
